// AMMAttention_76355928588528
// MI455X (gfx1250) — matrix operand order repaired and checked by reading, not compiled, not run on hardware
//
#include <hip/hip_runtime.h>

// ---------------------------------------------------------------------------
// Types for WMMA fragments (avoid HIP uint4: it has ctors and breaks unions)
// ---------------------------------------------------------------------------
typedef __attribute__((ext_vector_type(16))) __bf16       v16bf;
typedef __attribute__((ext_vector_type(8)))  float        v8f;
typedef __attribute__((ext_vector_type(4)))  unsigned int u32x4;

union Frag { v16bf bf; u32x4 u[2]; };

__device__ __forceinline__ unsigned short f2bf(float f) {
  unsigned u = __builtin_bit_cast(unsigned, f);
  return (unsigned short)((u + 0x7FFFu + ((u >> 16) & 1u)) >> 16); // RNE
}
__device__ __forceinline__ float bf2f(unsigned short h) {
  unsigned u = ((unsigned)h) << 16;
  return __builtin_bit_cast(float, u);
}
__device__ __forceinline__ v8f wmma_bf16(v16bf a, v16bf b, v8f c) {
  // D = A(16x32 bf16) * B(32x16 bf16) + C(16x16 f32)
  return __builtin_amdgcn_wmma_f32_16x16x32_bf16(false, a, false, b,
                                                 (short)0, c, false, false);
}
__device__ __forceinline__ v8f v8f_zero() {
  v8f z;
#pragma unroll
  for (int i = 0; i < 8; ++i) z[i] = 0.0f;
  return z;
}

// ---------------------------------------------------------------------------
// CDNA5 async global->LDS copy (ASYNCcnt-tracked, no VGPR round trip).
// LDS operand is the workgroup-relative byte offset (low 32 bits of the
// flat shared address); hardware adds the wave's LDS_BASE.
// ---------------------------------------------------------------------------
__device__ __forceinline__ void async_copy_b128(unsigned short* lds_dst,
                                                const unsigned short* gsrc) {
  unsigned loff = (unsigned)(size_t)(void*)lds_dst;
  unsigned long long gaddr = (unsigned long long)(size_t)gsrc;
  asm volatile("global_load_async_to_lds_b128 %0, %1, off"
               :: "v"(loff), "v"(gaddr) : "memory");
}
__device__ __forceinline__ void wait_async0() {
  asm volatile("s_wait_asynccnt 0x0" ::: "memory");
}

// Problem constants
#define MT    32768   // B*N rows
#define DK    512     // feature dim / GEMM K
#define QKVN  1536    // 3*512
#define NB    8
#define NSEQ  4096
#define NL    256     // landmarks
#define LDSK  48      // padded LDS K-stride (elems); 96B row, 16B aligned

// ---------------------------------------------------------------------------
// x (f32, M x 512) -> xb (bf16)
// ---------------------------------------------------------------------------
__global__ __launch_bounds__(256) void convert_x(const float* __restrict__ x,
                                                 unsigned short* __restrict__ xb) {
  size_t i = (size_t)blockIdx.x * 256 + threadIdx.x;  // one 8-float chunk
  const float* p = x + i * 8;
  float f[8];
#pragma unroll
  for (int j = 0; j < 8; ++j) f[j] = p[j];
  u32x4 o;
#pragma unroll
  for (int j = 0; j < 4; ++j)
    o[j] = (unsigned)f2bf(f[2 * j]) | ((unsigned)f2bf(f[2 * j + 1]) << 16);
  *(u32x4*)(xb + i * 8) = o;
}

// ---------------------------------------------------------------------------
// W (f32, K x Ncols) -> wT (bf16, Ncols x K)  (N-major, K contiguous)
// ---------------------------------------------------------------------------
__global__ __launch_bounds__(256) void transpose_w(const float* __restrict__ W,
                                                   unsigned short* __restrict__ wT,
                                                   int K, int Ncols) {
  int idx = blockIdx.x * 256 + threadIdx.x;  // grid sized exactly to K*Ncols
  int n = idx / K;
  int k = idx - n * K;
  wT[idx] = f2bf(W[(size_t)k * Ncols + n]);
}

// ---------------------------------------------------------------------------
// Fragment loaders
// ---------------------------------------------------------------------------
// B-fragment from LDS tile (lane = col, K {0-7,16-23}/{8-15,24-31} by half).
__device__ __forceinline__ void load_bfrag_lds(Frag& f, const unsigned short* Bs,
                                               int t, int l15, int half) {
  const unsigned short* p = &Bs[(t * 16 + l15) * LDSK + half * 8];
  f.u[0] = *(const u32x4*)(p);
  f.u[1] = *(const u32x4*)(p + 16);
}
// A-fragment from LDS tile (lane = row, K {0-7,16-23}/{8-15,24-31} by half).
__device__ __forceinline__ void load_afrag_lds(Frag& f, const unsigned short* As,
                                               int row, int half) {
  const unsigned short* p = &As[row * LDSK + half * 8];
  f.u[0] = *(const u32x4*)(p);
  f.u[1] = *(const u32x4*)(p + 16);
}

// ---------------------------------------------------------------------------
// Shared GEMM building blocks (128x128 tile, 8 waves, bf16 WMMA)
// ---------------------------------------------------------------------------
// Stage one 128x32 bf16 tile into LDS via async copies (256 threads).
__device__ __forceinline__ void stage_tile_async(unsigned short* lds,
                                                 const unsigned short* g,
                                                 int ld, int r0, int k0,
                                                 int tid) {
#pragma unroll
  for (int c = 0; c < 2; ++c) {                    // 512 b128 chunks total
    int chunk = tid * 2 + c;
    int row = chunk >> 2;
    int kc  = (chunk & 3) * 8;
    async_copy_b128(&lds[row * LDSK + kc],
                    g + (size_t)(r0 + row) * ld + k0 + kc);
  }
}

// One k-step: 3-stage software pipeline so 2 B-fragments stay in flight
// (forces distinct registers -> staggered s_wait_dscnt instead of 0x0).
__device__ __forceinline__ void compute_step(const unsigned short* As,
                                             const unsigned short* Bs,
                                             int wave, int l15, int half,
                                             v8f acc[8]) {
  Frag a;
  load_afrag_lds(a, As, wave * 16 + l15, half);
  Frag cur, nxt, nx2;
  load_bfrag_lds(cur, Bs, 0, l15, half);
  load_bfrag_lds(nxt, Bs, 1, l15, half);
#pragma unroll
  for (int t = 0; t < 8; ++t) {
    if (t + 2 < 8) load_bfrag_lds(nx2, Bs, t + 2, l15, half);
    acc[t] = wmma_bf16(a.bf, cur.bf, acc[t]);
    cur = nxt;
    nxt = nx2;
  }
}

// Double-buffered async mainloop over K=512.
__device__ __forceinline__ void gemm128_mainloop(
    const unsigned short* __restrict__ A, int lda, int m0,
    const unsigned short* __restrict__ Bt, int ldb, int n0,
    unsigned short* As0, unsigned short* As1,
    unsigned short* Bs0, unsigned short* Bs1,
    int tid, int wave, int l15, int half, v8f acc[8]) {
  stage_tile_async(As0, A, lda, m0, 0, tid);
  stage_tile_async(Bs0, Bt, ldb, n0, 0, tid);
  wait_async0();
  __syncthreads();
#pragma unroll 2
  for (int k0 = 0; k0 < DK; k0 += 32) {
    unsigned short* Ac = ((k0 >> 5) & 1) ? As1 : As0;
    unsigned short* Bc = ((k0 >> 5) & 1) ? Bs1 : Bs0;
    unsigned short* An = ((k0 >> 5) & 1) ? As0 : As1;
    unsigned short* Bn = ((k0 >> 5) & 1) ? Bs0 : Bs1;
    if (k0 + 32 < DK) {                 // prefetch next slice asynchronously
      stage_tile_async(An, A, lda, m0, k0 + 32, tid);
      stage_tile_async(Bn, Bt, ldb, n0, k0 + 32, tid);
    }
    compute_step(Ac, Bc, wave, l15, half, acc);
    wait_async0();
    __syncthreads();
  }
}

// ---------------------------------------------------------------------------
// QKV GEMM: qkvb[m, n] = xb[m, :] . wT[n, :]   (M=32768, N=1536, K=512)
// q (n<512) scaled by 1/sqrt(512).
// ---------------------------------------------------------------------------
__global__ __launch_bounds__(256) void gemm_qkv(
    const unsigned short* __restrict__ xb,
    const unsigned short* __restrict__ wT,
    unsigned short* __restrict__ qkvb) {
  __shared__ unsigned short As0[128 * LDSK], As1[128 * LDSK];
  __shared__ unsigned short Bs0[128 * LDSK], Bs1[128 * LDSK];
  const int tid  = threadIdx.x;
  const int lane = tid & 31;
  const int wave = tid >> 5;
  const int l15  = lane & 15;
  const int half = lane >> 4;
  const int m0 = blockIdx.x * 128;
  const int n0 = blockIdx.y * 128;

  v8f acc[8];
#pragma unroll
  for (int t = 0; t < 8; ++t) acc[t] = v8f_zero();

  gemm128_mainloop(xb, DK, m0, wT, DK, n0, As0, As1, Bs0, Bs1,
                   tid, wave, l15, half, acc);

  const float sc = (n0 < DK) ? 0.044194173824159216f : 1.0f;  // 1/sqrt(512)
#pragma unroll
  for (int t = 0; t < 8; ++t) {
    int gn = n0 + t * 16 + l15;
#pragma unroll
    for (int r = 0; r < 8; ++r) {        // C layout: M = r + 8*half, N = l15
      int gm = m0 + wave * 16 + half * 8 + r;
      qkvb[(size_t)gm * QKVN + gn] = f2bf(acc[t][r] * sc);
    }
  }
}

// ---------------------------------------------------------------------------
// Landmark pooling: mean over 16 consecutive rows.
// klm  (bf16, [b][l][d], row-major)  -> direct B-fragment source for Q.Klm^T
// vlmT (bf16, [b][d][l], transposed) -> direct B-fragment source for P.Vlm
// ---------------------------------------------------------------------------
__global__ __launch_bounds__(256) void pool_lm(
    const unsigned short* __restrict__ qkvb,
    unsigned short* __restrict__ klm,
    unsigned short* __restrict__ vlmT) {
  const int b = blockIdx.x >> 8;        // 0..7
  const int l = blockIdx.x & 255;       // 0..255
  const int d0 = threadIdx.x * 2;       // two features per thread
  const unsigned short* kbase =
      qkvb + (size_t)(b * NSEQ + l * 16) * QKVN + 512 + d0;
  const unsigned short* vbase = kbase + 512;
  float k0 = 0.f, k1 = 0.f, v0 = 0.f, v1 = 0.f;
#pragma unroll
  for (int s = 0; s < 16; ++s) {
    unsigned kk = *(const unsigned*)(kbase + (size_t)s * QKVN);
    unsigned vv = *(const unsigned*)(vbase + (size_t)s * QKVN);
    k0 += bf2f((unsigned short)(kk & 0xffff)); k1 += bf2f((unsigned short)(kk >> 16));
    v0 += bf2f((unsigned short)(vv & 0xffff)); v1 += bf2f((unsigned short)(vv >> 16));
  }
  const float inv = 1.0f / 16.0f;
  unsigned ko = (unsigned)f2bf(k0 * inv) | ((unsigned)f2bf(k1 * inv) << 16);
  *(unsigned*)(klm + (size_t)(b * NL + l) * DK + d0) = ko;
  vlmT[((size_t)b * DK + d0)     * NL + l] = f2bf(v0 * inv);
  vlmT[((size_t)b * DK + d0 + 1) * NL + l] = f2bf(v1 * inv);
}

// ---------------------------------------------------------------------------
// Attention: per WG = 64 q rows (4 waves x 16 rows).
// Phase 1: S(16x256) = q @ klm^T entirely in 128 accum VGPRs/lane; klm slices
//          double-buffered in LDS via async copies; B-fragments software-
//          pipelined 3-deep.
// Phase 2: softmax via __shfl_xor (width 16 matches C-layout M split).
// Phase 3: out = P @ vlm; P routed through per-wave LDS (aliases the klm
//          buffers, phase 1 is fenced off by its trailing barrier).
// ---------------------------------------------------------------------------
__global__ __launch_bounds__(128) void attn_kernel(
    const unsigned short* __restrict__ qkvb,
    const unsigned short* __restrict__ klm,
    const unsigned short* __restrict__ vlmT,
    unsigned short* __restrict__ ao) {
  __shared__ unsigned short smem[2 * NL * LDSK];   // 48 KB, dual-purpose
  unsigned short* Bsbuf0 = smem;
  unsigned short* Bsbuf1 = smem + NL * LDSK;
  const int tid  = threadIdx.x;
  const int lane = tid & 31;
  const int wave = tid >> 5;                   // 0..3
  const int l15  = lane & 15;
  const int half = lane >> 4;
  const int m0 = blockIdx.x * 64;              // global q row base
  const int b  = blockIdx.x >> 6;              // 64 blocks per batch
  const unsigned short* klmB = klm  + (size_t)b * NL * DK;
  const unsigned short* vlmB = vlmT + (size_t)b * DK * NL;

  v8f S[16];
#pragma unroll
  for (int t = 0; t < 16; ++t) S[t] = v8f_zero();

  const int arow = m0 + wave * 16 + l15;

  // ---- Phase 1: S = q @ klm^T -------------------------------------------
  {
#pragma unroll
    for (int c = 0; c < 8; ++c) {              // stage slice 0 (1024 b128)
      int chunk = tid * 8 + c;
      async_copy_b128(&Bsbuf0[(chunk >> 2) * LDSK + (chunk & 3) * 8],
                      klmB + (size_t)(chunk >> 2) * DK + (chunk & 3) * 8);
    }
    wait_async0();
    __syncthreads();
#pragma unroll 2
    for (int k0 = 0; k0 < DK; k0 += 32) {
      unsigned short* Bc = ((k0 >> 5) & 1) ? Bsbuf1 : Bsbuf0;
      unsigned short* Bn = ((k0 >> 5) & 1) ? Bsbuf0 : Bsbuf1;
      if (k0 + 32 < DK) {
#pragma unroll
        for (int c = 0; c < 8; ++c) {
          int chunk = tid * 8 + c;
          async_copy_b128(&Bn[(chunk >> 2) * LDSK + (chunk & 3) * 8],
                          klmB + (size_t)(chunk >> 2) * DK + k0 + 32 +
                              (chunk & 3) * 8);
        }
      }
      Frag a;  // q A-fragment straight from global (row stride 1536)
      const unsigned short* p = qkvb + (size_t)arow * QKVN + k0 + half * 8;
      a.u[0] = *(const u32x4*)(p);
      a.u[1] = *(const u32x4*)(p + 16);
      Frag cur, nxt, nx2;                      // 3-deep fragment pipeline
      load_bfrag_lds(cur, Bc, 0, l15, half);
      load_bfrag_lds(nxt, Bc, 1, l15, half);
#pragma unroll
      for (int t = 0; t < 16; ++t) {
        if (t + 2 < 16) load_bfrag_lds(nx2, Bc, t + 2, l15, half);
        S[t] = wmma_bf16(a.bf, cur.bf, S[t]);
        cur = nxt;
        nxt = nx2;
      }
      wait_async0();
      __syncthreads();
    }
  }

  // ---- Phase 2: softmax + write P into per-wave LDS (A-fragment layout) --
  // Aliases the klm staging buffers; safe after the loop's trailing barrier.
  unsigned short* P = smem + wave * (16 * 264);   // 16 rows x 264 (pad 8)
#pragma unroll
  for (int r = 0; r < 8; ++r) {
    float mx = -3.0e38f;
#pragma unroll
    for (int t = 0; t < 16; ++t) mx = fmaxf(mx, S[t][r]);
    mx = fmaxf(mx, __shfl_xor(mx, 1, 16));
    mx = fmaxf(mx, __shfl_xor(mx, 2, 16));
    mx = fmaxf(mx, __shfl_xor(mx, 4, 16));
    mx = fmaxf(mx, __shfl_xor(mx, 8, 16));
    float sum = 0.0f;
#pragma unroll
    for (int t = 0; t < 16; ++t) {
      float e = __expf(S[t][r] - mx);
      S[t][r] = e;
      sum += e;
    }
    sum += __shfl_xor(sum, 1, 16);
    sum += __shfl_xor(sum, 2, 16);
    sum += __shfl_xor(sum, 4, 16);
    sum += __shfl_xor(sum, 8, 16);
    float inv = 1.0f / sum;
    int prow = r + 8 * half;
#pragma unroll
    for (int t = 0; t < 16; ++t)
      P[prow * 264 + t * 16 + l15] = f2bf(S[t][r] * inv);
  }
  // Per-wave LDS region: same-wave DS ops are in-order; no barrier needed.

  // ---- Phase 3: out = P @ vlm -------------------------------------------
#pragma unroll 1
  for (int nc = 0; nc < DK; nc += 128) {       // output cols in 4 chunks
    v8f O[8];
#pragma unroll
    for (int t = 0; t < 8; ++t) O[t] = v8f_zero();
#pragma unroll 1
    for (int k0 = 0; k0 < NL; k0 += 32) {
      Frag a;                                  // P A-fragment from LDS
      const unsigned short* ap = &P[l15 * 264 + k0 + half * 8];
      a.u[0] = *(const u32x4*)(ap);
      a.u[1] = *(const u32x4*)(ap + 16);
      // 3-deep pipeline over global vlmT B-fragments (staggers loadcnt waits)
      Frag cur, nxt, nx2;
      {
        const unsigned short* bp =
            vlmB + (size_t)(nc + l15) * NL + k0 + half * 8;
        cur.u[0] = *(const u32x4*)(bp);
        cur.u[1] = *(const u32x4*)(bp + 16);
        bp += (size_t)16 * NL;
        nxt.u[0] = *(const u32x4*)(bp);
        nxt.u[1] = *(const u32x4*)(bp + 16);
      }
#pragma unroll
      for (int t = 0; t < 8; ++t) {
        if (t + 2 < 8) {
          const unsigned short* bp =
              vlmB + (size_t)(nc + (t + 2) * 16 + l15) * NL + k0 + half * 8;
          nx2.u[0] = *(const u32x4*)(bp);
          nx2.u[1] = *(const u32x4*)(bp + 16);
        }
        O[t] = wmma_bf16(a.bf, cur.bf, O[t]);
        cur = nxt;
        nxt = nx2;
      }
    }
#pragma unroll
    for (int t = 0; t < 8; ++t) {
      int gn = nc + t * 16 + l15;
#pragma unroll
      for (int r = 0; r < 8; ++r) {
        int gm = m0 + wave * 16 + half * 8 + r;
        ao[(size_t)gm * DK + gn] = f2bf(O[t][r]);
      }
    }
  }
}

// ---------------------------------------------------------------------------
// Projection + bias + residual:
// out[m, n] = ao[m, :] . wprojT[n, :] + b_proj[n] + v[m, n]   (f32 out)
// ---------------------------------------------------------------------------
__global__ __launch_bounds__(256) void gemm_proj(
    const unsigned short* __restrict__ ao,
    const unsigned short* __restrict__ wT,
    const float* __restrict__ bproj,
    const unsigned short* __restrict__ qkvb,
    float* __restrict__ out) {
  __shared__ unsigned short As0[128 * LDSK], As1[128 * LDSK];
  __shared__ unsigned short Bs0[128 * LDSK], Bs1[128 * LDSK];
  const int tid  = threadIdx.x;
  const int lane = tid & 31;
  const int wave = tid >> 5;
  const int l15  = lane & 15;
  const int half = lane >> 4;
  const int m0 = blockIdx.x * 128;
  const int n0 = blockIdx.y * 128;

  v8f acc[8];
#pragma unroll
  for (int t = 0; t < 8; ++t) acc[t] = v8f_zero();

  gemm128_mainloop(ao, DK, m0, wT, DK, n0, As0, As1, Bs0, Bs1,
                   tid, wave, l15, half, acc);

#pragma unroll
  for (int t = 0; t < 8; ++t) {
    int gn = n0 + t * 16 + l15;
    float bias = bproj[gn];
#pragma unroll
    for (int r = 0; r < 8; ++r) {
      int gm = m0 + wave * 16 + half * 8 + r;
      float resid = bf2f(qkvb[(size_t)gm * QKVN + 1024 + gn]);  // v residual
      out[(size_t)gm * DK + gn] = acc[t][r] + bias + resid;
    }
  }
}

// ---------------------------------------------------------------------------
// Launch. Workspace layout (bytes), total ~174 MB:
//   xb     @ 0          : 33,554,432   (x bf16, 32768 x 512)
//   qkvb   @ 33554432   : 100,663,296  (q|k|v bf16, 32768 x 1536, q pre-scaled)
//   wqkvT  @ 134217728  : 1,572,864    (bf16, 1536 x 512, N-major)
//   wprojT @ 135790592  : 524,288     (bf16, 512 x 512, N-major)
//   klm    @ 136314880  : 2,097,152   (bf16, 8 x 256 x 512)
//   vlmT   @ 138412032  : 2,097,152   (bf16, 8 x 512 x 256)
//   ao     @ 140509184  : 33,554,432  (bf16, 32768 x 512)
// ---------------------------------------------------------------------------
extern "C" void kernel_launch(void* const* d_in, const int* in_sizes, int n_in,
                              void* d_out, int out_size, void* d_ws, size_t ws_size,
                              hipStream_t stream) {
  const float* x      = (const float*)d_in[0];
  const float* W_qkv  = (const float*)d_in[1];
  const float* W_proj = (const float*)d_in[2];
  const float* b_proj = (const float*)d_in[3];
  float* out = (float*)d_out;

  char* w = (char*)d_ws;
  unsigned short* xb     = (unsigned short*)(w);
  unsigned short* qkvb   = (unsigned short*)(w + 33554432);
  unsigned short* wqkvT  = (unsigned short*)(w + 134217728);
  unsigned short* wprojT = (unsigned short*)(w + 135790592);
  unsigned short* klm    = (unsigned short*)(w + 136314880);
  unsigned short* vlmT   = (unsigned short*)(w + 138412032);
  unsigned short* ao     = (unsigned short*)(w + 140509184);

  convert_x  <<<8192, 256, 0, stream>>>(x, xb);                    // 2,097,152 chunks
  transpose_w<<<3072, 256, 0, stream>>>(W_qkv,  wqkvT,  512, 1536);
  transpose_w<<<1024, 256, 0, stream>>>(W_proj, wprojT, 512, 512);
  gemm_qkv   <<<dim3(256, 12), 256, 0, stream>>>(xb, wqkvT, qkvb);
  pool_lm    <<<2048, 256, 0, stream>>>(qkvb, klm, vlmT);
  attn_kernel<<<512, 128, 0, stream>>>(qkvb, klm, vlmT, ao);
  gemm_proj  <<<dim3(256, 4), 256, 0, stream>>>(ao, wprojT, b_proj, qkvb, out);
}
